// MechanismAttention_29205777613494
// MI455X (gfx1250) — matrix operand order repaired and checked by reading, not compiled, not run on hardware
//
#include <hip/hip_runtime.h>
#include <math.h>

// ---------------------------------------------------------------------------
// MechanismAttention for MI455X (gfx1250): bf16 WMMA GEMMs + fused attention
// with post-softmax mechanism gating. All matrix math goes through
// v_wmma_f32_16x16x32_bf16 (fp32 accumulate); softmax/LN/gate stay fp32.
// Attention uses a 128 KB fp32 score strip in LDS shared by 4 waves.
// Inner GEMM loops are explicitly software-pipelined (load k+1 before wmma k).
// ---------------------------------------------------------------------------

typedef __attribute__((ext_vector_type(16))) __bf16 v16bf;
typedef __attribute__((ext_vector_type(8)))  float  v8f;

#define DEV __device__ __forceinline__

static constexpr int Bsz   = 2;
static constexpr int Nseq  = 2048;   // T*S
static constexpr int Dm    = 512;
static constexpr int Hh    = 8;
static constexpr int DH    = 64;
static constexpr int INNER = 512;    // H*DH
static constexpr int Ssz   = 16;
static constexpr int Mrows = Bsz * Nseq;  // 4096 token rows

union BF16x16 { v16bf v; uint4 q[2]; };

DEV v8f wmma_bf16(v16bf a, v16bf b, v8f c) {
  return __builtin_amdgcn_wmma_f32_16x16x32_bf16(
      /*neg_a=*/false, a, /*neg_b=*/false, b,
      /*c_mod=*/(short)0, c, /*reuse_a=*/false, /*reuse_b=*/false);
}

// A-operand (16x32 bf16, row-major source, lane = row, split-half K chunks):
//   lane<16 : K = kbase+[0..7] and kbase+[16..23]
//   lane>=16: K = kbase+[8..15] and kbase+[24..31]
DEV v16bf load_a_g(const __bf16* __restrict__ base, int ld, int kbase, int lane) {
  int r = lane & 15, h = lane >> 4;
  const __bf16* p = base + (size_t)r * ld + kbase + h * 8;
  BF16x16 t;
  t.q[0] = *(const uint4*)(p);
  t.q[1] = *(const uint4*)(p + 16);
  return t.v;
}

// B-operand (32x16 bf16). Source is row-major (cols x K): lane = output column,
// split-half K chunks matching A: lane<16 K=[0..7],[16..23]; lane>=16 K=[8..15],[24..31].
DEV v16bf load_b_g(const __bf16* __restrict__ base, int ld, int col0, int kbase, int lane) {
  int c  = col0 + (lane & 15);
  int kb = kbase + (lane >> 4) * 8;
  const __bf16* p = base + (size_t)c * ld + kb;
  BF16x16 t;
  t.q[0] = *(const uint4*)(p);
  t.q[1] = *(const uint4*)(p + 16);
  return t.v;
}

// A-operand sourced from fp32 LDS (probability strip), converted to bf16.
DEV v16bf load_a_lds(const float* S, int ld, int kbase, int lane) {
  int r = lane & 15, h = lane >> 4;
  const float* p = S + r * ld + kbase + h * 8;
  float tmp[16];
  *(float4*)(tmp + 0)  = *(const float4*)(p + 0);
  *(float4*)(tmp + 4)  = *(const float4*)(p + 4);
  *(float4*)(tmp + 8)  = *(const float4*)(p + 16);
  *(float4*)(tmp + 12) = *(const float4*)(p + 20);
  v16bf a;
#pragma unroll
  for (int i = 0; i < 16; ++i) a[i] = (__bf16)tmp[i];
  return a;
}

DEV float wave_max(float v) {
#pragma unroll
  for (int o = 16; o; o >>= 1) v = fmaxf(v, __shfl_xor(v, o, 32));
  return v;
}
DEV float wave_sum(float v) {
#pragma unroll
  for (int o = 16; o; o >>= 1) v += __shfl_xor(v, o, 32);
  return v;
}

// ---------------------------------------------------------------------------
// Elementwise prep
// ---------------------------------------------------------------------------
__global__ void cast_bf16_kernel(const float* __restrict__ in, __bf16* __restrict__ out, int n) {
  for (int i = blockIdx.x * blockDim.x + threadIdx.x; i < n; i += gridDim.x * blockDim.x)
    out[i] = (__bf16)in[i];
}

__global__ void silu_bf16_kernel(const float* __restrict__ in, __bf16* __restrict__ out, int n) {
  for (int i = blockIdx.x * blockDim.x + threadIdx.x; i < n; i += gridDim.x * blockDim.x) {
    float x = in[i];
    out[i] = (__bf16)(x / (1.0f + expf(-x)));
  }
}

// One wave per token row: mean + rstd over D=512 (eps = 1e-5).
__global__ __launch_bounds__(32) void ln_stats_kernel(const float* __restrict__ x,
                                                      float* __restrict__ stats) {
  int row = blockIdx.x, lane = threadIdx.x;
  const float* p = x + (size_t)row * Dm;
  float s = 0.f;
  for (int i = lane; i < Dm; i += 32) s += p[i];
  float mu = wave_sum(s) * (1.0f / Dm);
  float v = 0.f;
  for (int i = lane; i < Dm; i += 32) { float d = p[i] - mu; v += d * d; }
  float rstd = rsqrtf(wave_sum(v) * (1.0f / Dm) + 1e-5f);
  if (lane == 0) { stats[2 * row] = mu; stats[2 * row + 1] = rstd; }
}

// xmod = bf16( LN(x) * (1+scale) + shift )
__global__ void apply_mod_kernel(const float* __restrict__ x, const float* __restrict__ stats,
                                 const float* __restrict__ lnw, const float* __restrict__ lnb,
                                 const float* __restrict__ mod, __bf16* __restrict__ xmod) {
  int total = Mrows * Dm;
  for (int i = blockIdx.x * blockDim.x + threadIdx.x; i < total; i += gridDim.x * blockDim.x) {
    int row = i >> 9, d = i & (Dm - 1);
    float mu = stats[2 * row], rstd = stats[2 * row + 1];
    float xn = (x[i] - mu) * rstd * lnw[d] + lnb[d];
    float sc = mod[(size_t)row * (2 * Dm) + d];
    float sh = mod[(size_t)row * (2 * Dm) + Dm + d];
    xmod[i] = (__bf16)(xn * (1.0f + sc) + sh);
  }
}

// ---------------------------------------------------------------------------
// Generic WMMA GEMM: C[M,Ncols] = A[M,K](bf16) @ W[Ncols,K](bf16)^T + bias
// grid = (M/16, Ncols/64), block = 32 (one wave -> 16x64 tile).
// Software-pipelined: operands for step k+32 are issued before the WMMAs of k.
// ---------------------------------------------------------------------------
__global__ __launch_bounds__(32) void gemm_bias_f32_kernel(const __bf16* __restrict__ A,
                                                           const __bf16* __restrict__ W,
                                                           const float* __restrict__ bias,
                                                           float* __restrict__ C,
                                                           int K, int Ncols) {
  int lane = threadIdx.x;
  int row0 = blockIdx.x * 16;
  int col0 = blockIdx.y * 64;
  const __bf16* Abase = A + (size_t)row0 * K;
  v8f acc[4] = {{}, {}, {}, {}};

  v16bf a_c = load_a_g(Abase, K, 0, lane);
  v16bf b_c0 = load_b_g(W, K, col0 +  0, 0, lane);
  v16bf b_c1 = load_b_g(W, K, col0 + 16, 0, lane);
  v16bf b_c2 = load_b_g(W, K, col0 + 32, 0, lane);
  v16bf b_c3 = load_b_g(W, K, col0 + 48, 0, lane);
  for (int k = 32; k < K; k += 32) {
    v16bf a_n = load_a_g(Abase, K, k, lane);
    v16bf b_n0 = load_b_g(W, K, col0 +  0, k, lane);
    v16bf b_n1 = load_b_g(W, K, col0 + 16, k, lane);
    v16bf b_n2 = load_b_g(W, K, col0 + 32, k, lane);
    v16bf b_n3 = load_b_g(W, K, col0 + 48, k, lane);
    __builtin_prefetch(Abase + (size_t)(lane & 15) * K + k + 32, 0, 3);
    acc[0] = wmma_bf16(a_c, b_c0, acc[0]);
    acc[1] = wmma_bf16(a_c, b_c1, acc[1]);
    acc[2] = wmma_bf16(a_c, b_c2, acc[2]);
    acc[3] = wmma_bf16(a_c, b_c3, acc[3]);
    a_c = a_n; b_c0 = b_n0; b_c1 = b_n1; b_c2 = b_n2; b_c3 = b_n3;
  }
  acc[0] = wmma_bf16(a_c, b_c0, acc[0]);
  acc[1] = wmma_bf16(a_c, b_c1, acc[1]);
  acc[2] = wmma_bf16(a_c, b_c2, acc[2]);
  acc[3] = wmma_bf16(a_c, b_c3, acc[3]);

  int cn = lane & 15, half = lane >> 4;
#pragma unroll
  for (int t = 0; t < 4; ++t) {
    int col = col0 + t * 16 + cn;
    float bv = bias ? bias[col] : 0.0f;
#pragma unroll
    for (int v = 0; v < 8; ++v) {
      int r = row0 + v + half * 8;
      C[(size_t)r * Ncols + col] = acc[t][v] + bv;
    }
  }
}

// QKV GEMM with head-scatter epilogue.
// Q (pre-scaled by 1/sqrt(DH)), K -> (b,h,n,dh) bf16 ; V -> transposed (b,h,dh,n).
__global__ __launch_bounds__(32) void gemm_qkv_kernel(const __bf16* __restrict__ A,
                                                      const __bf16* __restrict__ W,
                                                      __bf16* __restrict__ q,
                                                      __bf16* __restrict__ kk,
                                                      __bf16* __restrict__ vT) {
  const int K = Dm;
  int lane = threadIdx.x;
  int row0 = blockIdx.x * 16;
  int col0 = blockIdx.y * 64;
  const __bf16* Abase = A + (size_t)row0 * K;
  v8f acc[4] = {{}, {}, {}, {}};

  v16bf a_c = load_a_g(Abase, K, 0, lane);
  v16bf b_c0 = load_b_g(W, K, col0 +  0, 0, lane);
  v16bf b_c1 = load_b_g(W, K, col0 + 16, 0, lane);
  v16bf b_c2 = load_b_g(W, K, col0 + 32, 0, lane);
  v16bf b_c3 = load_b_g(W, K, col0 + 48, 0, lane);
  for (int k = 32; k < K; k += 32) {
    v16bf a_n = load_a_g(Abase, K, k, lane);
    v16bf b_n0 = load_b_g(W, K, col0 +  0, k, lane);
    v16bf b_n1 = load_b_g(W, K, col0 + 16, k, lane);
    v16bf b_n2 = load_b_g(W, K, col0 + 32, k, lane);
    v16bf b_n3 = load_b_g(W, K, col0 + 48, k, lane);
    acc[0] = wmma_bf16(a_c, b_c0, acc[0]);
    acc[1] = wmma_bf16(a_c, b_c1, acc[1]);
    acc[2] = wmma_bf16(a_c, b_c2, acc[2]);
    acc[3] = wmma_bf16(a_c, b_c3, acc[3]);
    a_c = a_n; b_c0 = b_n0; b_c1 = b_n1; b_c2 = b_n2; b_c3 = b_n3;
  }
  acc[0] = wmma_bf16(a_c, b_c0, acc[0]);
  acc[1] = wmma_bf16(a_c, b_c1, acc[1]);
  acc[2] = wmma_bf16(a_c, b_c2, acc[2]);
  acc[3] = wmma_bf16(a_c, b_c3, acc[3]);

  int cn = lane & 15, half = lane >> 4;
#pragma unroll
  for (int t = 0; t < 4; ++t) {
    int col = col0 + t * 16 + cn;            // 0..1535
    int which = col / INNER;                 // 0=q 1=k 2=v
    int within = col - which * INNER;
    int h = within >> 6, dh = within & 63;
#pragma unroll
    for (int v = 0; v < 8; ++v) {
      int row = row0 + v + half * 8;         // 0..4095
      int b = row >> 11, n = row & (Nseq - 1);
      int bh = b * Hh + h;
      float fv = acc[t][v];
      if (which == 0)      q [((size_t)bh * Nseq + n) * DH + dh] = (__bf16)(fv * 0.125f);
      else if (which == 1) kk[((size_t)bh * Nseq + n) * DH + dh] = (__bf16)fv;
      else                 vT[((size_t)bh * DH + dh) * Nseq + n] = (__bf16)fv;
    }
  }
}

// ---------------------------------------------------------------------------
// Mechanism gate: gate[b,h,i,j] = sigmoid( gelu(m_ij @ g1^T + b1) @ g2^T + b2 )
// One block (256 thr) per (b,i,j) row of m_ij; row staged in LDS once.
// ---------------------------------------------------------------------------
__global__ __launch_bounds__(256) void gate_kernel(const float* __restrict__ m_ij,
                                                   const float* __restrict__ g1w,
                                                   const float* __restrict__ g1b,
                                                   const float* __restrict__ g2w,
                                                   const float* __restrict__ g2b,
                                                   float* __restrict__ gate) {
  __shared__ float srow[Dm];
  __shared__ float h1[256];
  int r = blockIdx.x;          // 0..511 = (b*16+i)*16+j
  int c = threadIdx.x;         // hidden unit 0..255
  const float* mrow = m_ij + (size_t)r * Dm;
  for (int i = c; i < Dm; i += 256) srow[i] = mrow[i];
  __syncthreads();
  const float* wrow = g1w + (size_t)c * Dm;
  float s = g1b[c];
  for (int i = 0; i < Dm; ++i) s += srow[i] * wrow[i];
  h1[c] = 0.5f * s * (1.0f + erff(s * 0.70710678118654752f));
  __syncthreads();
  if (c < Hh) {
    const float* w2 = g2w + (size_t)c * 256;
    float t = g2b[c];
    for (int i = 0; i < 256; ++i) t += h1[i] * w2[i];
    float sg = 1.0f / (1.0f + expf(-t));
    int b = r >> 8, ij = r & 255;
    gate[((size_t)(b * Hh + c)) * 256 + ij] = sg;   // (b,h,i,j)
  }
}

// ---------------------------------------------------------------------------
// Attention: one block (4 waves) per (b,h, 16-query tile). Full 16x2048 fp32
// score strip in LDS (128 KB of the WGP's 320 KB), shared by the 4 waves:
//   wave w: key tiles j == w (mod 4) -> softmax rows 4w..4w+3 -> P@V col tile w
// Q comes pre-scaled by 1/sqrt(DH). Post-softmax mechanism gate.
// Both WMMA loops are software-pipelined on their global operand.
// ---------------------------------------------------------------------------
__global__ __launch_bounds__(128) void attn_kernel(const __bf16* __restrict__ q,
                                                   const __bf16* __restrict__ kk,
                                                   const __bf16* __restrict__ vT,
                                                   const float* __restrict__ gate,
                                                   __bf16* __restrict__ o) {
  __shared__ float sS[16 * Nseq];          // 131072 bytes
  int tid = threadIdx.x;
  int lane = tid & 31;
  int w = tid >> 5;                        // wave id 0..3
  int qt = blockIdx.x & 127;               // query tile (N/16)
  int bh = blockIdx.x >> 7;                // 0..15
  const __bf16* qbase = q  + ((size_t)bh * Nseq + (size_t)qt * 16) * DH;
  const __bf16* kbase = kk + (size_t)bh * Nseq * DH;
  const __bf16* vbase = vT + (size_t)bh * DH * Nseq;
  int cn = lane & 15, half = lane >> 4;

  // Q tile as two A-operands (K = dh 0..31 and 32..63); each wave keeps a copy.
  v16bf aq0 = load_a_g(qbase, DH, 0, lane);
  v16bf aq1 = load_a_g(qbase, DH, 32, lane);

  // S = (Q/sqrt(DH)) K^T into LDS; wave w handles key tiles j = w, w+4, ...
  {
    int j = w;
    v16bf b0 = load_b_g(kbase, DH, j * 16, 0, lane);
    v16bf b1 = load_b_g(kbase, DH, j * 16, 32, lane);
    for (; j + 4 < Nseq / 16; j += 4) {
      v16bf n0 = load_b_g(kbase, DH, (j + 4) * 16, 0, lane);
      v16bf n1 = load_b_g(kbase, DH, (j + 4) * 16, 32, lane);
      __builtin_prefetch(kbase + (size_t)(j + 8) * 16 * DH, 0, 3);
      v8f c = {};
      c = wmma_bf16(aq0, b0, c);
      c = wmma_bf16(aq1, b1, c);
#pragma unroll
      for (int v = 0; v < 8; ++v)
        sS[(v + half * 8) * Nseq + j * 16 + cn] = c[v];
      b0 = n0; b1 = n1;
    }
    v8f c = {};
    c = wmma_bf16(aq0, b0, c);
    c = wmma_bf16(aq1, b1, c);
#pragma unroll
    for (int v = 0; v < 8; ++v)
      sS[(v + half * 8) * Nseq + j * 16 + cn] = c[v];
  }
  __syncthreads();

  // Row-wise softmax then post-softmax mechanism gate; wave w owns rows 4w..4w+3.
  // n % 16 == local row r; m % 16 == lane % 16 when striding by 32.
  for (int r = w * 4; r < w * 4 + 4; ++r) {
    float* Srow = sS + r * Nseq;
    float mx = -1e30f;
    for (int m = lane; m < Nseq; m += 32) mx = fmaxf(mx, Srow[m]);
    mx = wave_max(mx);
    float sum = 0.f;
    for (int m = lane; m < Nseq; m += 32) {
      float e = expf(Srow[m] - mx);
      Srow[m] = e;
      sum += e;
    }
    sum = wave_sum(sum);
    float gv = gate[(size_t)bh * 256 + r * 16 + (lane & 15)];
    float scl = gv / sum;
    for (int m = lane; m < Nseq; m += 32) Srow[m] *= scl;
  }
  __syncthreads();

  // O = P @ V : wave w computes dh column tile w (16 cols) over all keys.
  v8f acc = {};
  {
    v16bf bv = load_b_g(vbase, Nseq, w * 16, 0, lane);
    int j = 0;
    for (; j + 1 < Nseq / 32; ++j) {
      v16bf bn = load_b_g(vbase, Nseq, w * 16, (j + 1) * 32, lane);
      v16bf a = load_a_lds(sS, Nseq, j * 32, lane);
      acc = wmma_bf16(a, bv, acc);
      bv = bn;
    }
    v16bf a = load_a_lds(sS, Nseq, j * 32, lane);
    acc = wmma_bf16(a, bv, acc);
  }
  int b_ = bh >> 3, h_ = bh & 7;
  int dh = w * 16 + cn;
#pragma unroll
  for (int v = 0; v < 8; ++v) {
    int n = qt * 16 + v + half * 8;
    o[(((size_t)(b_ * Nseq + n)) * Hh + h_) * DH + dh] = (__bf16)acc[v];
  }
}

// ---------------------------------------------------------------------------
// Host-side orchestration
// ---------------------------------------------------------------------------
extern "C" void kernel_launch(void* const* d_in, const int* in_sizes, int n_in,
                              void* d_out, int out_size, void* d_ws, size_t ws_size,
                              hipStream_t stream) {
  (void)in_sizes; (void)n_in; (void)out_size; (void)ws_size;
  const float* x     = (const float*)d_in[0];
  const float* m_ij  = (const float*)d_in[1];
  const float* aemb  = (const float*)d_in[2];
  const float* lnw   = (const float*)d_in[3];
  const float* lnb   = (const float*)d_in[4];
  const float* modw  = (const float*)d_in[5];
  const float* modb  = (const float*)d_in[6];
  const float* qkvw  = (const float*)d_in[7];
  const float* outw  = (const float*)d_in[8];
  const float* outb  = (const float*)d_in[9];
  const float* g1w   = (const float*)d_in[10];
  const float* g1b   = (const float*)d_in[11];
  const float* g2w   = (const float*)d_in[12];
  const float* g2b   = (const float*)d_in[13];
  float* out = (float*)d_out;

  char* ws = (char*)d_ws;
  size_t off = 0;
  auto alloc = [&](size_t bytes) { char* p = ws + off; off += (bytes + 255) & ~size_t(255); return p; };
  float*  w_mod   = (float*)alloc((size_t)Mrows * 2 * Dm * 4);      // 16 MB scale||shift
  float*  w_stats = (float*)alloc((size_t)Mrows * 2 * 4);
  __bf16* w_a     = (__bf16*)alloc((size_t)Mrows * Dm * 2);         // silu(action)
  __bf16* w_xmod  = (__bf16*)alloc((size_t)Mrows * Dm * 2);
  __bf16* w_q     = (__bf16*)alloc((size_t)Bsz * Hh * Nseq * DH * 2);
  __bf16* w_k     = (__bf16*)alloc((size_t)Bsz * Hh * Nseq * DH * 2);
  __bf16* w_vT    = (__bf16*)alloc((size_t)Bsz * Hh * Nseq * DH * 2);
  __bf16* w_o     = (__bf16*)alloc((size_t)Mrows * INNER * 2);
  __bf16* w_modw  = (__bf16*)alloc((size_t)2 * Dm * Dm * 2);
  __bf16* w_qkvw  = (__bf16*)alloc((size_t)3 * INNER * Dm * 2);
  __bf16* w_outw  = (__bf16*)alloc((size_t)Dm * INNER * 2);
  float*  w_gate  = (float*)alloc((size_t)Bsz * Hh * Ssz * Ssz * 4);

  // 1) weight casts + silu(action) cast
  {
    int n1 = 2 * Dm * Dm, n2 = 3 * INNER * Dm, n3 = Dm * INNER, n4 = Mrows * Dm;
    cast_bf16_kernel<<<(n1 + 255) / 256, 256, 0, stream>>>(modw, w_modw, n1);
    cast_bf16_kernel<<<(n2 + 255) / 256, 256, 0, stream>>>(qkvw, w_qkvw, n2);
    cast_bf16_kernel<<<(n3 + 255) / 256, 256, 0, stream>>>(outw, w_outw, n3);
    silu_bf16_kernel<<<(n4 + 255) / 256, 256, 0, stream>>>(aemb, w_a, n4);
  }
  // 2) LayerNorm stats
  ln_stats_kernel<<<Mrows, 32, 0, stream>>>(x, w_stats);
  // 3) modulation GEMM: (4096x512) @ (1024x512)^T
  gemm_bias_f32_kernel<<<dim3(Mrows / 16, (2 * Dm) / 64), 32, 0, stream>>>(
      w_a, w_modw, modb, w_mod, Dm, 2 * Dm);
  // 4) apply LN + AdaLN
  apply_mod_kernel<<<(Mrows * Dm + 255) / 256, 256, 0, stream>>>(
      x, w_stats, lnw, lnb, w_mod, w_xmod);
  // 5) QKV GEMM: (4096x512) @ (1536x512)^T with head scatter (Q pre-scaled)
  gemm_qkv_kernel<<<dim3(Mrows / 16, (3 * INNER) / 64), 32, 0, stream>>>(
      w_xmod, w_qkvw, w_q, w_k, w_vT);
  // 6) mechanism gate (small MLP)
  gate_kernel<<<Bsz * Ssz * Ssz, 256, 0, stream>>>(m_ij, g1w, g1b, g2w, g2b, w_gate);
  // 7) gated attention, 4 waves per (b,h,16-query strip)
  attn_kernel<<<Bsz * Hh * (Nseq / 16), 128, 0, stream>>>(w_q, w_k, w_vT, w_gate, w_o);
  // 8) output projection: (4096x512) @ (512x512)^T + out_b -> fp32 d_out
  gemm_bias_f32_kernel<<<dim3(Mrows / 16, Dm / 64), 32, 0, stream>>>(
      w_o, w_outw, outb, out, INNER, Dm);
}
